// sparse_conv2d_naive_74483322847783
// MI455X (gfx1250) — matrix operand order repaired and checked by reading, not compiled, not run on hardware
//
#include <hip/hip_runtime.h>

typedef _Float16 v16h __attribute__((ext_vector_type(16)));
typedef _Float16 v8h  __attribute__((ext_vector_type(8)));
typedef _Float16 h4   __attribute__((ext_vector_type(4)));
typedef float    v8f  __attribute__((ext_vector_type(8)));

#define Nn   4
#define Cc   64
#define Hh   128
#define Ww   128
#define OCn  64
#define KNn  9
#define Kt   576      // C*KN (contraction length)
#define HWn  16384    // H*W
#define TILE 32       // spatial positions per block
#define RS   584      // LDS col row stride in halfs (576 + 8 pad)

// ---------- kernel 1: x NCHW -> NHWC (per batch, LDS tiled transpose) ----------
__global__ __launch_bounds__(256) void xpose_kernel(const float* __restrict__ x,
                                                    float* __restrict__ xt) {
  __shared__ float t[32][33];
  const int n  = blockIdx.z;
  const int c0 = blockIdx.y * 32;
  const int p0 = blockIdx.x * 32;       // hw tile base
  const int tx = threadIdx.x;           // 0..31
  const int ty = threadIdx.y;           // 0..7
  const float* xs = x  + (size_t)n * Cc * HWn;
  float*       xd = xt + (size_t)n * HWn * Cc;
#pragma unroll
  for (int i = 0; i < 4; ++i)           // read coalesced along hw
    t[ty + i * 8][tx] = xs[(size_t)(c0 + ty + i * 8) * HWn + p0 + tx];
  __syncthreads();
#pragma unroll
  for (int i = 0; i < 4; ++i)           // write coalesced along c
    xd[(size_t)(p0 + ty + i * 8) * Cc + c0 + tx] = t[tx][ty + i * 8];
}

// ---------- kernel 2: weight f32 (OC,C,KN) -> f16 [oc][kn*64 + c] ----------
__global__ __launch_bounds__(256) void wprep_kernel(const float* __restrict__ w,
                                                    _Float16* __restrict__ wb) {
  const int idx = blockIdx.x * 256 + threadIdx.x;
  if (idx >= OCn * Kt) return;
  const int oc = idx / Kt;
  const int r  = idx - oc * Kt;   // r = kn*64 + c
  const int kn = r >> 6;
  const int c  = r & 63;
  wb[idx] = (_Float16)w[(oc * Cc + c) * KNn + kn];
}

// ---------- kernel 3: fused deformable im2col (LDS, f16) + WMMA GEMM ----------
__global__ __launch_bounds__(256) void dcn_kernel(const float* __restrict__ xt,
                                                  const float* __restrict__ off,
                                                  const _Float16* __restrict__ wb,
                                                  float* __restrict__ out) {
  __shared__ _Float16 col[TILE * RS];   // 32 * 584 * 2 = 37376 bytes

  const int bx      = blockIdx.x;
  const int n       = bx >> 9;          // 512 blocks per batch
  const int posBase = (bx & 511) * TILE;
  const int tid     = threadIdx.x;

  // ===== phase 1: bilinear sampling into LDS col tile =====
  {
    const int grp = tid >> 4;           // 16 groups of 16 lanes
    const int l16 = tid & 15;           // each lane handles 4 channels
    const float* offn = off + (size_t)n * (2 * KNn) * HWn;
    const float* xtn  = xt  + (size_t)n * HWn * Cc;
#pragma unroll 1
    for (int it = 0; it < (TILE * KNn) / 16; ++it) {   // 18 iterations
      const int p   = it * 16 + grp;    // (pos, kn) pair id, 0..287
      const int pos = p / 9;
      const int kn  = p - pos * 9;
      const int hw  = posBase + pos;
      const int oh  = hw >> 7;
      const int ow  = hw & 127;
      // base collapses to the center pixel for every tap (see reference)
      const float fy = (float)oh + offn[(2 * kn + 0) * HWn + hw];
      const float fx = (float)ow + offn[(2 * kn + 1) * HWn + hw];
      const float y0f = floorf(fy), x0f = floorf(fx);
      const int y0 = (int)y0f, x0 = (int)x0f;
      const int y1 = y0 + 1,   x1 = x0 + 1;
      const float wy1 = fy - y0f, wy0 = 1.0f - wy1;
      const float wx1 = fx - x0f, wx0 = 1.0f - wx1;
      const float my0 = (y0 >= 0 && y0 <= Hh - 1) ? 1.0f : 0.0f;
      const float my1 = (y1 >= 0 && y1 <= Hh - 1) ? 1.0f : 0.0f;
      const float mx0 = (x0 >= 0 && x0 <= Ww - 1) ? 1.0f : 0.0f;
      const float mx1 = (x1 >= 0 && x1 <= Ww - 1) ? 1.0f : 0.0f;
      const float w00 = wx0 * wy0 * mx0 * my0;
      const float w10 = wx1 * wy0 * mx1 * my0;
      const float w01 = wx0 * wy1 * mx0 * my1;
      const float w11 = wx1 * wy1 * mx1 * my1;
      const int y0c = min(max(y0, 0), Hh - 1), y1c = min(max(y1, 0), Hh - 1);
      const int x0c = min(max(x0, 0), Ww - 1), x1c = min(max(x1, 0), Ww - 1);
      const float4 v00 = *((const float4*)(xtn + (size_t)((y0c * Ww + x0c) * Cc)) + l16);
      const float4 v10 = *((const float4*)(xtn + (size_t)((y0c * Ww + x1c) * Cc)) + l16);
      const float4 v01 = *((const float4*)(xtn + (size_t)((y1c * Ww + x0c) * Cc)) + l16);
      const float4 v11 = *((const float4*)(xtn + (size_t)((y1c * Ww + x1c) * Cc)) + l16);
      h4 r;
      r.x = (_Float16)(w00 * v00.x + w10 * v10.x + w01 * v01.x + w11 * v11.x);
      r.y = (_Float16)(w00 * v00.y + w10 * v10.y + w01 * v01.y + w11 * v11.y);
      r.z = (_Float16)(w00 * v00.z + w10 * v10.z + w01 * v01.z + w11 * v11.z);
      r.w = (_Float16)(w00 * v00.w + w10 * v10.w + w01 * v01.w + w11 * v11.w);
      *(h4*)(&col[pos * RS + (kn << 6) + (l16 << 2)]) = r;   // 8B aligned ds_store
    }
  }
  __syncthreads();

  // ===== phase 2: WMMA GEMM  out[64 x 32tile] = W[64 x 576] * col[576 x 32] =====
  const int wave = tid >> 5;            // 8 waves
  const int lane = tid & 31;
  const int l    = lane & 15;
  const int g    = lane >> 4;
  const int ni   = wave & 1;            // N tile (2 of 16 cols)
  const int mi   = wave >> 1;           // M tile (4 of 16 rows)

  v8f acc = {};
  const _Float16* arow  = wb + (size_t)(mi * 16 + l) * Kt;         // A: row = M
  const _Float16* bbase = &col[(ni * 16 + l) * RS];                // B: row = N col

#pragma unroll
  for (int kb = 0; kb < Kt / 32; ++kb) {                           // 18 WMMA steps
    // A fragment (16-bit A 16x32 layout): h<8 -> K=8g+h ; h>=8 -> K=16+8g+(h-8)
    const v8h alo = *(const v8h*)(arow + kb * 32 + 8 * g);
    const v8h ahi = *(const v8h*)(arow + kb * 32 + 16 + 8 * g);
    // B fragment (16-bit B 32x16 layout): h<8 -> K=8g+h ; h>=8 -> K=16+8g+(h-8), col = lane%16
    const v8h blo = *(const v8h*)(bbase + kb * 32 + 8 * g);
    const v8h bhi = *(const v8h*)(bbase + kb * 32 + 16 + 8 * g);
    v16h a, b;
#pragma unroll
    for (int i = 0; i < 8; ++i) {
      a[i] = alo[i]; a[i + 8] = ahi[i];
      b[i] = blo[i]; b[i + 8] = bhi[i];
    }
    acc = __builtin_amdgcn_wmma_f32_16x16x32_f16(
        /*neg_a=*/false, a, /*neg_b=*/false, b,
        /*c_mod=*/(short)0, acc, /*reuse_a=*/false, /*reuse_b=*/false);
  }

  // D layout: VGPR j -> M = j + 8g, N = lane%16
  const int hw     = posBase + ni * 16 + l;
  const int ocBase = mi * 16 + 8 * g;
  float* outn = out + (size_t)n * OCn * HWn;
#pragma unroll
  for (int j = 0; j < 8; ++j)
    outn[(size_t)(ocBase + j) * HWn + hw] = acc[j];
}

extern "C" void kernel_launch(void* const* d_in, const int* in_sizes, int n_in,
                              void* d_out, int out_size, void* d_ws, size_t ws_size,
                              hipStream_t stream) {
  const float* x   = (const float*)d_in[0];   // (4,64,128,128) f32
  const float* off = (const float*)d_in[1];   // (4,18,128,128) f32
  const float* w   = (const float*)d_in[2];   // (64,64,9)      f32
  float* out = (float*)d_out;                 // (4,64,128,128) f32

  float*    xt = (float*)d_ws;                                        // 16 MB NHWC copy
  _Float16* wb = (_Float16*)((char*)d_ws +
                             (size_t)Nn * HWn * Cc * sizeof(float));  // 72 KB f16 weights

  dim3 tb(32, 8);
  dim3 tg(HWn / 32, Cc / 32, Nn);
  xpose_kernel<<<tg, tb, 0, stream>>>(x, xt);
  wprep_kernel<<<(OCn * Kt + 255) / 256, 256, 0, stream>>>(w, wb);
  dcn_kernel<<<Nn * (HWn / TILE), 256, 0, stream>>>(xt, off, wb, out);
}
